// VectorMemoryStore_20229295964724
// MI455X (gfx1250) — matrix operand order repaired and checked by reading, not compiled, not run on hardware
//
#include <hip/hip_runtime.h>
#include <hip/hip_bf16.h>

constexpr int HS = 512;     // hidden
constexpr int KS = 512;     // key
constexpr int VS = 512;     // value
constexpr int MS = 16384;   // memory slots
constexpr int TS = 1024;    // B*S tokens

typedef __attribute__((ext_vector_type(16))) _Float16 v16h;
typedef __attribute__((ext_vector_type(8)))  _Float16 v8h;
typedef __attribute__((ext_vector_type(8)))  float    v8f;
typedef __attribute__((ext_vector_type(4)))  int      v4i;

// ---- CDNA5 async global->LDS path (guarded; sync LDS copy fallback) ----
#ifndef USE_ASYNC_LDS
#if defined(__has_builtin)
#  if __has_builtin(__builtin_amdgcn_global_load_async_to_lds_b128) && \
      __has_builtin(__builtin_amdgcn_s_wait_asynccnt)
#    define USE_ASYNC_LDS 1
#  else
#    define USE_ASYNC_LDS 0
#  endif
#else
#  define USE_ASYNC_LDS 0
#endif
#endif

__device__ __forceinline__ void lds_fill16(const _Float16* g, _Float16* l) {
#if USE_ASYNC_LDS
  // generic->AS1 and generic->AS3 via integer casts (LDS offset = low 32 bits).
  // Builtin prototype (from compiler diagnostic): (AS1 v4i*, AS3 v4i*, imm, imm)
  __builtin_amdgcn_global_load_async_to_lds_b128(
      (__attribute__((address_space(1))) v4i*)(uintptr_t)g,
      (__attribute__((address_space(3))) v4i*)(unsigned)(uintptr_t)l,
      0, 0);
#else
  *(v8h*)l = *(const v8h*)g;
#endif
}
__device__ __forceinline__ void lds_fill_wait() {
#if USE_ASYNC_LDS
  __builtin_amdgcn_s_wait_asynccnt(0);
#endif
}

// ---- ordered-float packing for argmax/argmin with first-index tie-break ----
__device__ __forceinline__ unsigned f2ord(float f) {
  unsigned u = __float_as_uint(f);
  return (u & 0x80000000u) ? ~u : (u | 0x80000000u);
}
__device__ __forceinline__ float ord2f(unsigned k) {
  unsigned u = (k & 0x80000000u) ? (k ^ 0x80000000u) : ~k;
  return __uint_as_float(u);
}

// =======================================================================
// WMMA GEMM:  C[M,N] = A[M,K] * B[N,K]^T (+bias[N]), f16 in, f32 out.
// Block = 8 waves; block tile = 16(M) x 512(N); wave tile = 16x64
// (4 accumulators, A fragment reused 4x). A k-chunk (16x32 f16 = 1KB)
// is staged in double-buffered LDS, filled with async global->LDS loads
// overlapping the WMMA burst. M%16==0, N%512==0, K%32==0 at all sites.
// =======================================================================
__global__ __launch_bounds__(256) void wmma_gemm_nt(
    const _Float16* __restrict__ A,
    const _Float16* __restrict__ Bm,
    const float*    __restrict__ bias,
    float*          __restrict__ C,
    _Float16*       __restrict__ Ch,
    int M, int N, int K)
{
  __shared__ _Float16 abuf[2][16 * 32];
  const int tid  = threadIdx.x;
  const int lane = tid & 31;
  const int wave = tid >> 5;
  const int m0 = blockIdx.y << 4;
  const int n0 = (blockIdx.x << 9) + (wave << 6);
  const int hi = lane >> 4;        // half-wave selector
  const int r  = lane & 15;

  // B fragment (32x16): lane holds col r; halves 0..15 = K[16*hi .. 16*hi+15]
  const _Float16* b0 = Bm + (size_t)(n0 + r) * K + (hi << 4);
  const size_t bstr = (size_t)16 * K;     // next 16-column group

  // A stage: 64 filler lanes, 16B each, cover 16 rows x 32 halves
  const int frow = tid >> 2, fseg = tid & 3;
  const _Float16* gsrc = A + (size_t)(m0 + frow) * K + (fseg << 3);

  v8f acc0 = {}, acc1 = {}, acc2 = {}, acc3 = {};

  if (tid < 64) lds_fill16(gsrc, &abuf[0][frow * 32 + (fseg << 3)]);
  lds_fill_wait();
  __syncthreads();

  const int iters = K >> 5;
  for (int i = 0; i < iters; ++i) {
    const int cur = i & 1;
    if (i + 1 < iters && tid < 64)
      lds_fill16(gsrc + ((i + 1) << 5), &abuf[cur ^ 1][frow * 32 + (fseg << 3)]);

    // A fragment (16x32): lane r; halves 0..15 = K[16*hi .. 16*hi+15]
    const _Float16* ab = &abuf[cur][r * 32 + (hi << 4)];
    v8h  alo = *(const v8h*)(ab);
    v8h  ahi = *(const v8h*)(ab + 8);
    v16h a = __builtin_shufflevector(alo, ahi,
               0,1,2,3,4,5,6,7,8,9,10,11,12,13,14,15);
    const _Float16* bkp = b0 + ((size_t)i << 5);
    v16h bb0 = *(const v16h*)(bkp);
    v16h bb1 = *(const v16h*)(bkp + bstr);
    v16h bb2 = *(const v16h*)(bkp + 2 * bstr);
    v16h bb3 = *(const v16h*)(bkp + 3 * bstr);
    acc0 = __builtin_amdgcn_wmma_f32_16x16x32_f16(false, a, false, bb0, (short)0, acc0, false, false);
    acc1 = __builtin_amdgcn_wmma_f32_16x16x32_f16(false, a, false, bb1, (short)0, acc1, false, false);
    acc2 = __builtin_amdgcn_wmma_f32_16x16x32_f16(false, a, false, bb2, (short)0, acc2, false, false);
    acc3 = __builtin_amdgcn_wmma_f32_16x16x32_f16(false, a, false, bb3, (short)0, acc3, false, false);

    if (((i & 3) == 0) && (i + 8 < iters)) {   // one cacheline ahead per stream
      __builtin_prefetch((const void*)(bkp + 256), 0, 1);
      __builtin_prefetch((const void*)(bkp + bstr + 256), 0, 1);
      __builtin_prefetch((const void*)(bkp + 2 * bstr + 256), 0, 1);
      __builtin_prefetch((const void*)(bkp + 3 * bstr + 256), 0, 1);
    }
    lds_fill_wait();        // next A chunk landed (overlapped with WMMAs)
    __syncthreads();
  }

  // C/D: lane<16 -> c[i]=D[i][lane]; lane>=16 -> c[i]=D[8+i][lane-16]
  const int rbase = m0 + (hi << 3);
#define STORE_TILE(ACC, T)                                            \
  {                                                                   \
    const int col = n0 + ((T) << 4) + r;                              \
    const float bv = bias ? bias[col] : 0.0f;                         \
    _Pragma("unroll")                                                 \
    for (int i = 0; i < 8; ++i) {                                     \
      float v = ACC[i] + bv;                                          \
      C[(size_t)(rbase + i) * N + col] = v;                           \
      if (Ch) Ch[(size_t)(rbase + i) * N + col] = (_Float16)v;        \
    }                                                                 \
  }
  STORE_TILE(acc0, 0)
  STORE_TILE(acc1, 1)
  STORE_TILE(acc2, 2)
  STORE_TILE(acc3, 3)
#undef STORE_TILE
}

// ---- elementwise f32 -> f16 ----
__global__ void cvt_f16(const float* __restrict__ in, _Float16* __restrict__ out, int n) {
  int i = blockIdx.x * blockDim.x + threadIdx.x;
  if (i < n) out[i] = (_Float16)in[i];
}

// ---- mvT[v][m] = (f16) mv[m][v]  (so attn @ MV fits the A*B^T GEMM form) ----
__global__ void transpose_f16(const float* __restrict__ mv, _Float16* __restrict__ mvT) {
  int i = blockIdx.x * blockDim.x + threadIdx.x;   // over VS*MS outputs
  int v = i >> 14;            // / MS
  int m = i & (MS - 1);
  mvT[(size_t)v * MS + m] = (_Float16)mv[(size_t)m * VS + v];
}

// ---- normalize memory keys: mkn_h = l2norm rows (f16), inv_cur = 1/max(||row||,eps) ----
__global__ __launch_bounds__(256) void prep_memkeys(
    const float* __restrict__ mk, _Float16* __restrict__ mkn_h,
    float* __restrict__ inv_cur)
{
  const int lane = threadIdx.x & 31;
  const int row  = blockIdx.x * 8 + (threadIdx.x >> 5);
  const float* src = mk + (size_t)row * KS;
  float ss = 0.f;
  for (int j = lane; j < KS; j += 32) { float v = src[j]; ss += v * v; }
  for (int m = 16; m; m >>= 1) ss += __shfl_xor(ss, m, 32);
  float inv = 1.0f / fmaxf(sqrtf(ss), 1e-12f);
  for (int j = lane; j < KS; j += 32)
    mkn_h[(size_t)row * KS + j] = (_Float16)(src[j] * inv);
  if (lane == 0) inv_cur[row] = inv;
}

// ---- row-normalize query keys -> f32 + f16 copies ----
__global__ __launch_bounds__(256) void normalize_qk(
    const float* __restrict__ qkraw, float* __restrict__ qk, _Float16* __restrict__ qkh)
{
  const int lane = threadIdx.x & 31;
  const int row  = blockIdx.x * 8 + (threadIdx.x >> 5);
  const float* src = qkraw + (size_t)row * KS;
  float ss = 0.f;
  for (int j = lane; j < KS; j += 32) { float v = src[j]; ss += v * v; }
  for (int m = 16; m; m >>= 1) ss += __shfl_xor(ss, m, 32);
  float inv = 1.0f / fmaxf(sqrtf(ss), 1e-12f);
  for (int j = lane; j < KS; j += 32) {
    float v = src[j] * inv;
    qk [(size_t)row * KS + j] = v;
    qkh[(size_t)row * KS + j] = (_Float16)v;
  }
}

// ---- masked softmax over M per token row; attn in f16 ----
__global__ __launch_bounds__(256) void softmax_rows(
    const float* __restrict__ S, const float* __restrict__ usage,
    _Float16* __restrict__ attn)
{
  __shared__ float red[256];
  const int row = blockIdx.x, tid = threadIdx.x;
  const float* s = S + (size_t)row * MS;
  float mx = -3.0e38f;
  for (int j = tid; j < MS; j += 256) {
    float v = (usage[j] > 0.f) ? s[j] * 10.0f : -1.0e9f;
    mx = fmaxf(mx, v);
  }
  red[tid] = mx; __syncthreads();
  for (int o = 128; o; o >>= 1) { if (tid < o) red[tid] = fmaxf(red[tid], red[tid + o]); __syncthreads(); }
  mx = red[0]; __syncthreads();
  float sum = 0.f;
  for (int j = tid; j < MS; j += 256) {
    float v = (usage[j] > 0.f) ? s[j] * 10.0f : -1.0e9f;
    sum += __expf(v - mx);
  }
  red[tid] = sum; __syncthreads();
  for (int o = 128; o; o >>= 1) { if (tid < o) red[tid] += red[tid + o]; __syncthreads(); }
  float inv = 1.0f / red[0];
  for (int j = tid; j < MS; j += 256) {
    float v = (usage[j] > 0.f) ? s[j] * 10.0f : -1.0e9f;
    attn[(size_t)row * MS + j] = (_Float16)(__expf(v - mx) * inv);
  }
}

// =======================================================================
// Sequential upsert/evict scan: single persistent workgroup, 1024 threads.
// Baseline similarity comes from the precomputed raw-sims matrix; only rows
// modified so far ("dirty") get a fresh f32 dot / maintained inverse norm.
// =======================================================================
__global__ __launch_bounds__(1024) void scan_update(
    const float* __restrict__ sraw,   // [TS, MS] raw dots vs pre-update memory
    const float* __restrict__ qk,     // [TS, KS]
    const float* __restrict__ vals,   // [TS, VS]
    float* __restrict__ mk,           // [MS, KS] (output copy, updated in place)
    float* __restrict__ mv,           // [MS, VS]
    float* __restrict__ mu,           // [MS]
    float* __restrict__ inv_cur,      // [MS] working 1/max(norm,eps)
    unsigned char* __restrict__ dirty,// [MS] zeroed before launch
    int T)
{
  __shared__ unsigned long long sMax, sMin;
  __shared__ int sIdx, sUpd;
  __shared__ float sred[1024];
  const int tid = threadIdx.x;

  for (int t = 0; t < T; ++t) {
    const float* kt   = qk   + (size_t)t * KS;
    const float* vt   = vals + (size_t)t * VS;
    const float* srow = sraw + (size_t)t * MS;
    if (tid == 0) { sMax = 0ull; sMin = ~0ull; }
    __syncthreads();

    unsigned long long lMax = 0ull, lMin = ~0ull;
    for (int m = tid; m < MS; m += 1024) {
      float s;
      if (dirty[m]) {
        const float* row = mk + (size_t)m * KS;
        float d = 0.f;
#pragma unroll 8
        for (int j = 0; j < KS; ++j) d += row[j] * kt[j];
        s = d * inv_cur[m];
      } else {
        s = srow[m];
      }
      unsigned long long pk =
          ((unsigned long long)f2ord(s) << 32) | (unsigned)(0xFFFFFFFFu - (unsigned)m);
      if (pk > lMax) lMax = pk;
      unsigned long long pu =
          ((unsigned long long)f2ord(mu[m]) << 32) | (unsigned)m;
      if (pu < lMin) lMin = pu;
    }
    atomicMax(&sMax, lMax);
    atomicMin(&sMin, lMin);
    __syncthreads();

    if (tid == 0) {
      unsigned long long pm = sMax;
      float maxSim = ord2f((unsigned)(pm >> 32));
      int maxIdx = (int)(0xFFFFFFFFu - (unsigned)(pm & 0xFFFFFFFFu));
      int upd = (maxSim > 0.7f) && (mu[maxIdx] > 0.f);
      int idx = upd ? maxIdx : (int)(sMin & 0xFFFFFFFFu);
      sIdx = idx; sUpd = upd;
      mu[idx] = upd ? (mu[idx] + 1.0f) : 1.0f;
      dirty[idx] = 1;
    }
    __syncthreads();

    const int idx = sIdx, upd = sUpd;
    float* krow = mk + (size_t)idx * KS;
    float* vrow = mv + (size_t)idx * VS;
    if (tid < KS) {
      float nk = upd ? (0.95f * krow[tid] + 0.05f * kt[tid]) : kt[tid];
      krow[tid] = nk;
      sred[tid] = nk * nk;
    } else {
      int j = tid - KS;
      float nv = upd ? (0.95f * vrow[j] + 0.05f * vt[j]) : vt[j];
      vrow[j] = nv;
      sred[tid] = 0.f;
    }
    __syncthreads();
    for (int o = 256; o; o >>= 1) {
      if (tid < o) sred[tid] += sred[tid + o];
      __syncthreads();
    }
    if (tid == 0)
      inv_cur[idx] = 1.0f / fmaxf(sqrtf(sred[0]), 1e-12f);
    __syncthreads();
  }
}

extern "C" void kernel_launch(void* const* d_in, const int* in_sizes, int n_in,
                              void* d_out, int out_size, void* d_ws, size_t ws_size,
                              hipStream_t stream) {
  (void)in_sizes; (void)n_in; (void)out_size; (void)ws_size;
  const float* hidden = (const float*)d_in[0];
  const float* Wk     = (const float*)d_in[1];
  const float* bk     = (const float*)d_in[2];
  const float* Wv     = (const float*)d_in[3];
  const float* bvv    = (const float*)d_in[4];
  const float* Wo     = (const float*)d_in[5];
  const float* bo     = (const float*)d_in[6];
  const float* mk_in  = (const float*)d_in[7];
  const float* mv_in  = (const float*)d_in[8];
  const float* mu_in  = (const float*)d_in[9];

  float* out    = (float*)d_out;                     // [TS, HS]
  float* out_mk = out    + (size_t)TS * HS;          // [MS, KS]
  float* out_mv = out_mk + (size_t)MS * KS;          // [MS, VS]
  float* out_mu = out_mv + (size_t)MS * VS;          // [MS]

  char* ws = (char*)d_ws;
  size_t off = 0;
  auto carve = [&](size_t bytes) -> void* {
    void* p = ws + off;
    off = (off + bytes + 255) & ~(size_t)255;
    return p;
  };
  _Float16* hidden_h = (_Float16*)carve((size_t)TS * HS * 2);
  _Float16* Wk_h     = (_Float16*)carve((size_t)KS * HS * 2);
  _Float16* Wv_h     = (_Float16*)carve((size_t)VS * HS * 2);
  _Float16* Wo_h     = (_Float16*)carve((size_t)HS * VS * 2);
  _Float16* mkn_h    = (_Float16*)carve((size_t)MS * KS * 2);
  _Float16* mvT_h    = (_Float16*)carve((size_t)VS * MS * 2);
  float*    inv_cur  = (float*)   carve((size_t)MS * 4);
  unsigned char* dirty = (unsigned char*)carve((size_t)MS);
  float*    qkraw    = (float*)   carve((size_t)TS * KS * 4);
  float*    qk_f     = (float*)   carve((size_t)TS * KS * 4);
  _Float16* qk_h     = (_Float16*)carve((size_t)TS * KS * 2);
  float*    vals_f   = (float*)   carve((size_t)TS * VS * 4);
  float*    sraw     = (float*)   carve((size_t)TS * MS * 4);
  _Float16* attn_h   = (_Float16*)carve((size_t)TS * MS * 2);
  float*    retr_f   = (float*)   carve((size_t)TS * VS * 4);
  _Float16* retr_h   = (_Float16*)carve((size_t)TS * VS * 2);

  // ---- precision staging ----
  int n;
  n = TS * HS; cvt_f16<<<(n + 255) / 256, 256, 0, stream>>>(hidden, hidden_h, n);
  n = KS * HS; cvt_f16<<<(n + 255) / 256, 256, 0, stream>>>(Wk, Wk_h, n);
  n = VS * HS; cvt_f16<<<(n + 255) / 256, 256, 0, stream>>>(Wv, Wv_h, n);
  n = HS * VS; cvt_f16<<<(n + 255) / 256, 256, 0, stream>>>(Wo, Wo_h, n);
  prep_memkeys<<<MS / 8, 256, 0, stream>>>(mk_in, mkn_h, inv_cur);
  n = VS * MS; transpose_f16<<<n / 256, 256, 0, stream>>>(mv_in, mvT_h);

  // ---- read path (all WMMA) ----
  wmma_gemm_nt<<<dim3(KS / 512, TS / 16), 256, 0, stream>>>(
      hidden_h, Wk_h, bk, qkraw, nullptr, TS, KS, HS);
  normalize_qk<<<TS / 8, 256, 0, stream>>>(qkraw, qk_f, qk_h);
  wmma_gemm_nt<<<dim3(VS / 512, TS / 16), 256, 0, stream>>>(
      hidden_h, Wv_h, bvv, vals_f, nullptr, TS, VS, HS);
  wmma_gemm_nt<<<dim3(MS / 512, TS / 16), 256, 0, stream>>>(
      qk_h, mkn_h, nullptr, sraw, nullptr, TS, MS, KS);      // raw dots (pre-temperature)
  softmax_rows<<<TS, 256, 0, stream>>>(sraw, mu_in, attn_h);
  wmma_gemm_nt<<<dim3(VS / 512, TS / 16), 256, 0, stream>>>(
      attn_h, mvT_h, nullptr, retr_f, retr_h, TS, VS, MS);
  wmma_gemm_nt<<<dim3(HS / 512, TS / 16), 256, 0, stream>>>(
      retr_h, Wo_h, bo, out, nullptr, TS, HS, VS);

  // ---- write path: copy memory into output, then persistent scan updates it ----
  (void)hipMemcpyAsync(out_mk, mk_in, (size_t)MS * KS * 4, hipMemcpyDeviceToDevice, stream);
  (void)hipMemcpyAsync(out_mv, mv_in, (size_t)MS * VS * 4, hipMemcpyDeviceToDevice, stream);
  (void)hipMemcpyAsync(out_mu, mu_in, (size_t)MS * 4,      hipMemcpyDeviceToDevice, stream);
  (void)hipMemsetAsync(dirty, 0, MS, stream);
  scan_update<<<1, 1024, 0, stream>>>(sraw, qk_f, vals_f, out_mk, out_mv, out_mu,
                                      inv_cur, dirty, TS);
}
